// VQ_VAE_81604378624349
// MI455X (gfx1250) — compile-verified, matrix operand order checked by reading, not run on hardware
//
#include <hip/hip_runtime.h>
#include <hip/hip_bf16.h>
#include <stdint.h>

typedef __attribute__((ext_vector_type(16))) __bf16 v16bf;
typedef __attribute__((ext_vector_type(8)))  __bf16 v8bf;
typedef __attribute__((ext_vector_type(8)))  float  v8f;

#define B_  32
#define T_  2048
#define D_  512
#define K_  1024
#define N_  (B_*T_)              // 65536 rows
#define ROWS_PER_BLOCK 128       // 8 waves x 16 rows
#define WAVES 8
#define THREADS 256
#define CTP_TILES (K_/32)        // 32 iterations of 2 codeword tiles (32 codewords)
#define DT_TILES (D_/32)         // 16 k-chunks of 32

// workspace layout (bytes)
#define WS_LOSS_OFF   0                      // 1 float
#define WS_EBF16_OFF  16                     // K_*D_ bf16 = 1 MB
#define WS_ENORM_OFF  (16 + K_*D_*2)         // K_ floats

// dynamic LDS: double-buffered B staging only
#define B_ROW_BYTES 1040                     // 512 halves + 8 pad halves (bank shift)
#define BUF_BYTES   (32 * B_ROW_BYTES)       // 33280 per buffer (32 codeword rows)
#define LDS_TOTAL   (2 * BUF_BYTES)          // 66560

__device__ __forceinline__ unsigned f2bf_bits(float f) {
    unsigned u = __float_as_uint(f);
    u += 0x7FFFu + ((u >> 16) & 1u);   // round-to-nearest-even
    return u >> 16;
}
__device__ __forceinline__ unsigned pack2(float a, float b) {
    return f2bf_bits(a) | (f2bf_bits(b) << 16);
}

// build a 16-half fragment from two 16B chunks (ISA K-split layout)
__device__ __forceinline__ v16bf ldfrag(const char* p) {
    union { uint4 u; v8bf h; } lo, hi;
    lo.u = *(const uint4*)(p);
    hi.u = *(const uint4*)(p + 32);
    v16bf r;
#pragma unroll
    for (int i = 0; i < 8; ++i) { r[i] = lo.h[i]; r[i + 8] = hi.h[i]; }
    return r;
}

// async global->LDS stage of one 32-codeword bf16 tile (2048 x 16B chunks)
__device__ __forceinline__ void stage_async(const unsigned short* __restrict__ ebf,
                                            char* smem, int pair, int buf, int t) {
    const int k0 = pair * 32;
#pragma unroll
    for (int j = 0; j < 8; ++j) {
        int c    = t + j * 256;          // 0..2047, coalesced per j
        int brow = c >> 6;
        int off  = c & 63;
        unsigned lds_off = (unsigned)(uintptr_t)(smem + buf * BUF_BYTES
                                                 + brow * B_ROW_BYTES + off * 16);
        unsigned voff    = (unsigned)(((k0 + brow) << 10) + off * 16); // bytes into ebf
        asm volatile("global_load_async_to_lds_b128 %0, %1, %2"
                     :: "v"(lds_off), "v"(voff), "s"(ebf) : "memory");
    }
}
__device__ __forceinline__ void wait_async0() {
    asm volatile("s_wait_asynccnt 0x0" ::: "memory");
}

// ---------------------------------------------------------------------------
// Kernel 1: codebook prep. One wave per codeword row:
//   enorm[k] = ||e_k||^2 ; bf16 copy of embedding ; zero loss accumulator.
// ---------------------------------------------------------------------------
__global__ void __launch_bounds__(256) vq_prep(const float* __restrict__ emb,
                                               float* __restrict__ ws_loss,
                                               unsigned short* __restrict__ ebf,
                                               float* __restrict__ enorm) {
    if (blockIdx.x == 0 && threadIdx.x == 0) *ws_loss = 0.0f;
    int wave = (blockIdx.x * blockDim.x + threadIdx.x) >> 5;
    int lane = threadIdx.x & 31;
    if (wave >= K_) return;
    const float* row = emb + (size_t)wave * D_;
    unsigned* out = (unsigned*)(ebf + (size_t)wave * D_);
    float s = 0.f;
#pragma unroll
    for (int i = 0; i < 4; ++i) {
        int d = lane * 4 + i * 128;
        float4 v = *(const float4*)(row + d);
        s += v.x*v.x + v.y*v.y + v.z*v.z + v.w*v.w;
        out[d/2]   = pack2(v.x, v.y);
        out[d/2+1] = pack2(v.z, v.w);
    }
#pragma unroll
    for (int off = 16; off; off >>= 1) s += __shfl_xor(s, off, 32);
    if (lane == 0) enorm[wave] = s;
}

// ---------------------------------------------------------------------------
// Kernel 2: A-stationary bf16 WMMA distance GEMM + argmin + gather + loss.
// Wave holds its 16x512 input tile in 16 v16bf fragments (128 VGPRs).
// B tiles double-buffered in LDS via async global->LDS loads.
// ---------------------------------------------------------------------------
__global__ void __launch_bounds__(THREADS) vq_main(
        const float* __restrict__ x, const float* __restrict__ emb,
        const unsigned short* __restrict__ ebf, const float* __restrict__ enorm,
        float* __restrict__ out_q, float* __restrict__ out_idx,
        float* __restrict__ ws_loss) {
    extern __shared__ char smem[];
    __shared__ int   s_bestk[WAVES][16];
    __shared__ float s_loss;

    const int t    = threadIdx.x;
    const int wave = t >> 5;
    const int lane = t & 31;
    const int m    = lane & 15;   // A row-in-tile / B col-in-tile / C: n
    const int sub  = lane >> 4;   // half-wave K sub-range select (ISA layout)
    const int row0_block = blockIdx.x * ROWS_PER_BLOCK;

    // kick off first B tile while we convert A
    stage_async(ebf, smem, 0, 0, t);
    if (t == 0) s_loss = 0.f;

    // ---- load this wave's A tile into registers as bf16 fragments ----
    // lane (m,sub) per dt holds X[row][d0 + {sub*8..+7, 16+sub*8..+7}]
    v16bf afr[DT_TILES];
    {
        const float* xr = x + (size_t)(row0_block + wave * 16 + m) * D_;
#pragma unroll
        for (int dt = 0; dt < DT_TILES; ++dt) {
            const float* p = xr + dt * 32 + sub * 8;
            float4 q0 = *(const float4*)(p);
            float4 q1 = *(const float4*)(p + 4);
            float4 q2 = *(const float4*)(p + 16);
            float4 q3 = *(const float4*)(p + 20);
            union { unsigned u[8]; v16bf v; } a;
            a.u[0] = pack2(q0.x, q0.y); a.u[1] = pack2(q0.z, q0.w);
            a.u[2] = pack2(q1.x, q1.y); a.u[3] = pack2(q1.z, q1.w);
            a.u[4] = pack2(q2.x, q2.y); a.u[5] = pack2(q2.z, q2.w);
            a.u[6] = pack2(q3.x, q3.y); a.u[7] = pack2(q3.z, q3.w);
            afr[dt] = a.v;
        }
    }

    float bestv[8];
    int   bestk[8];
#pragma unroll
    for (int v = 0; v < 8; ++v) { bestv[v] = 3.4e38f; bestk[v] = 0; }

    wait_async0();
    __syncthreads();

    for (int ctp = 0; ctp < CTP_TILES; ++ctp) {
        const int cur = ctp & 1;
        const int k0  = ctp * 32;
        if (ctp + 1 < CTP_TILES) stage_async(ebf, smem, ctp + 1, cur ^ 1, t);

        float en0 = enorm[k0 + m];        // overlap with compute below
        float en1 = enorm[k0 + 16 + m];

        v8f acc0 = {}, acc1 = {};
        const char* base = smem + cur * BUF_BYTES;
#pragma unroll
        for (int dt = 0; dt < DT_TILES; ++dt) {
            const int boff = (dt * 32 + sub * 8) * 2;
            v16bf b0 = ldfrag(base + m * B_ROW_BYTES + boff);
            v16bf b1 = ldfrag(base + (16 + m) * B_ROW_BYTES + boff);
            acc0 = __builtin_amdgcn_wmma_f32_16x16x32_bf16(
                       false, afr[dt], false, b0, (short)0, acc0, false, false);
            acc1 = __builtin_amdgcn_wmma_f32_16x16x32_bf16(
                       false, afr[dt], false, b1, (short)0, acc1, false, false);
        }
        // score = ||e_k||^2 - 2 x.e_k  (||x||^2 dropped: constant per row)
#pragma unroll
        for (int v = 0; v < 8; ++v) {
            float s0 = __builtin_fmaf(-2.0f, acc0[v], en0);
            float s1 = __builtin_fmaf(-2.0f, acc1[v], en1);
            if (s0 < bestv[v]) { bestv[v] = s0; bestk[v] = k0 + m; }
            if (s1 < bestv[v]) { bestv[v] = s1; bestk[v] = k0 + 16 + m; }
        }
        wait_async0();
        __syncthreads();
    }

    // ---- argmin across the 16 lanes of each half-wave (n dimension) ----
#pragma unroll
    for (int v = 0; v < 8; ++v) {
        float bv = bestv[v]; int bk = bestk[v];
#pragma unroll
        for (int off = 8; off; off >>= 1) {
            float ov = __shfl_xor(bv, off, 16);
            int   ok = __shfl_xor(bk, off, 16);
            if (ov < bv || (ov == bv && ok < bk)) { bv = ov; bk = ok; }
        }
        bestv[v] = bv; bestk[v] = bk;
    }
    if (m == 0) {   // lane 0 -> rows v ; lane 16 -> rows v+8
#pragma unroll
        for (int v = 0; v < 8; ++v) s_bestk[wave][v + 8 * sub] = bestk[v];
    }
    __syncthreads();

    // ---- gather winning codewords (coalesced), outputs + loss partial ----
    float lsum = 0.f;
    for (int r = 0; r < 16; ++r) {
        int k = s_bestk[wave][r];
        size_t grow = (size_t)(row0_block + wave * 16 + r);
        const float* er = emb + (size_t)k * D_;
        const float* xr = x + grow * D_;
        float*       qr = out_q + grow * D_;
        if (lane == 0) out_idx[grow] = (float)k;
#pragma unroll
        for (int i = 0; i < 4; ++i) {
            int c = (lane + i * 32) * 4;
            float4 q  = *(const float4*)(er + c);
            float4 xv = *(const float4*)(xr + c);
            *(float4*)(qr + c) = q;      // quantized_st == quantized numerically
            float dx = q.x - xv.x, dy = q.y - xv.y;
            float dz = q.z - xv.z, dw = q.w - xv.w;
            lsum += dx*dx + dy*dy + dz*dz + dw*dw;
        }
    }
    atomicAdd(&s_loss, lsum);           // ds_add_f32
    __syncthreads();
    if (t == 0) atomicAdd(ws_loss, s_loss);
}

// ---------------------------------------------------------------------------
// Kernel 3: finalize scalar losses.
// ---------------------------------------------------------------------------
__global__ void vq_finish(const float* __restrict__ ws_loss,
                          float* __restrict__ out_losses) {
    float mse = *ws_loss * (1.0f / 33554432.0f);   // / (B*T*D)
    out_losses[0] = 0.25f * mse;   // vq_loss (COMMITMENT_COST)
    out_losses[1] = mse;           // e_latent_loss
}

extern "C" void kernel_launch(void* const* d_in, const int* in_sizes, int n_in,
                              void* d_out, int out_size, void* d_ws, size_t ws_size,
                              hipStream_t stream) {
    const float* x   = (const float*)d_in[0];   // [32,2048,512] f32
    const float* emb = (const float*)d_in[1];   // [1024,512]    f32
    float* out = (float*)d_out;                 // quantized | idx | vq_loss | e_latent
    char*  ws  = (char*)d_ws;

    float*          ws_loss = (float*)(ws + WS_LOSS_OFF);
    unsigned short* ebf     = (unsigned short*)(ws + WS_EBF16_OFF);
    float*          enormp  = (float*)(ws + WS_ENORM_OFF);

    vq_prep<<<K_/WAVES, 256, 0, stream>>>(emb, ws_loss, ebf, enormp);
    vq_main<<<N_/ROWS_PER_BLOCK, THREADS, LDS_TOTAL, stream>>>(
        x, emb, ebf, enormp, out, out + (size_t)N_ * D_, ws_loss);
    vq_finish<<<1, 1, 0, stream>>>(ws_loss, out + (size_t)N_ * D_ + N_);
}
